// DecoderRNN_91036126806867
// MI455X (gfx1250) — matrix operand order repaired and checked by reading, not compiled, not run on hardware
//
#include <hip/hip_runtime.h>
#include <stdint.h>

// ---------------- Types ----------------
typedef __attribute__((ext_vector_type(16))) _Float16 v16h;
typedef __attribute__((ext_vector_type(8)))  _Float16 v8h;
typedef __attribute__((ext_vector_type(8)))  float    v8f;
typedef __attribute__((ext_vector_type(4)))  unsigned int v4u;
typedef __attribute__((ext_vector_type(8)))  int      v8i;
typedef __attribute__((ext_vector_type(4)))  int      v4i;

#define B_TOT   32768
#define LATENT  128
#define H       100
#define HP      112          // H padded to 7x16 for WMMA N tiles
#define KP      128          // K padded to 4x32 for WMMA K steps
#define VOCAB   20
#define TSTEPS  21
#define ROWS    128          // batch rows per workgroup (8 waves x 16)

// ws layout (in halves):
//   [0)              wproj padded  [112][128]            = 14336
//   [14336)          6 big mats (wih0,whh0,wih1,whh1,wih2,whh2) each [336][128] = 43008
//   [14336+6*43008)  wout padded   [32][128]             = 4096
#define WS_WPROJ_OFF  0
#define WS_BIG_OFF    14336
#define WS_BIG_ONE    43008
#define WS_WOUT_OFF   (14336 + 6*43008)

// ---------------- Weight prep: f32 -> padded f16 ----------------
__global__ void prep_weights(const float* __restrict__ wproj,
                             const float* __restrict__ wih0, const float* __restrict__ whh0,
                             const float* __restrict__ wih1, const float* __restrict__ whh1,
                             const float* __restrict__ wih2, const float* __restrict__ whh2,
                             const float* __restrict__ wout,
                             _Float16* __restrict__ ws) {
  int idx = blockIdx.x * blockDim.x + threadIdx.x;
  const int WPROJ_N = HP * KP;          // 14336
  const int BIG_N   = 3 * HP * KP;      // 43008
  const int WOUT_N  = 32 * KP;          // 4096
  const int total = WPROJ_N + 6 * BIG_N + WOUT_N;
  if (idx >= total) return;
  float v = 0.f;
  if (idx < WPROJ_N) {                       // w_proj [100][128] -> [112][128]
    int r = idx >> 7, c = idx & 127;
    if (r < H) v = wproj[r * LATENT + c];
    ws[WS_WPROJ_OFF + idx] = (_Float16)v;
    return;
  }
  idx -= WPROJ_N;
  if (idx < 6 * BIG_N) {                     // [300][100] -> [3][112][128]
    int m = idx / BIG_N, e = idx % BIG_N;
    int r = e >> 7, c = e & 127;
    int gate = r / HP, n = r % HP;
    const float* src = (m == 0) ? wih0 : (m == 1) ? whh0 : (m == 2) ? wih1
                     : (m == 3) ? whh1 : (m == 4) ? wih2 : whh2;
    if (n < H && c < H) v = src[(gate * H + n) * H + c];
    ws[WS_BIG_OFF + idx] = (_Float16)v;
    return;
  }
  idx -= 6 * BIG_N;                          // w_out [20][100] -> [32][128]
  int r = idx >> 7, c = idx & 127;
  if (r < VOCAB && c < H) v = wout[r * H + c];
  ws[WS_WOUT_OFF + idx] = (_Float16)v;
}

// ---------------- LDS image ----------------
struct SMem {
  _Float16 wbuf[2 * 3 * HP * KP];   // 86016 halves = 172032 B  (wih | whh of current layer)
  _Float16 x0[ROWS * KP];           // 32768 B
  _Float16 h[3][ROWS * KP];         // 98304 B
  _Float16 wout[32 * KP];           // 8192 B
  float    bias[2160];              // bproj[112] | 3x(bih[336],bhh[336]) | bout[32] = 8640 B
};
static_assert(sizeof(SMem) == 172032 + 32768 + 98304 + 8192 + 8640, "smem layout");

// ---------------- WMMA fragment helpers ----------------
union U16 { v16h v; v8h h[2]; };

__device__ __forceinline__ v16h loadA(const _Float16* base, int m0, int s, int lm, int hi) {
  // 16-bit A 16x32: lane holds K=[k0,k0+8) and K=[k0+16,k0+24), k0 = 32*s + (hi?8:0)
  int off = (m0 + lm) * KP + 32 * s + (hi ? 8 : 0);
  U16 u;
  u.h[0] = *reinterpret_cast<const v8h*>(base + off);
  u.h[1] = *reinterpret_cast<const v8h*>(base + off + 16);
  return u.v;
}
__device__ __forceinline__ v16h loadB(const _Float16* base, int n0, int s, int lm, int hi) {
  // 16-bit B 32x16: lane = column N, holds K=[k0,k0+8) and K=[k0+16,k0+24), k0 = 32*s + (hi?8:0)
  int off = (n0 + lm) * KP + 32 * s + (hi ? 8 : 0);
  U16 u;
  u.h[0] = *reinterpret_cast<const v8h*>(base + off);
  u.h[1] = *reinterpret_cast<const v8h*>(base + off + 16);
  return u.v;
}
__device__ __forceinline__ v8f wmma16(v16h a, v16h b, v8f c) {
  return __builtin_amdgcn_wmma_f32_16x16x32_f16(false, a, false, b, (short)0, c, false, false);
}
__device__ __forceinline__ v8f vzero8() {
  v8f z;
#pragma unroll
  for (int i = 0; i < 8; i++) z[i] = 0.f;
  return z;
}
__device__ __forceinline__ float sigm(float x) { return 1.0f / (1.0f + __expf(-x)); }
__device__ __forceinline__ float tanh_f(float x) {
  x = fminf(15.f, fmaxf(-15.f, x));
  float e = __expf(2.0f * x);
  return (e - 1.0f) / (e + 1.0f);
}

// ---------------- TDM stage: global -> LDS (2D qword tile, D# per ISA 8.3-8.5) ----------------
__device__ __forceinline__ void tdm_issue(unsigned lds_addr, unsigned long long gaddr,
                                          unsigned cols_q, unsigned rows) {
  v4u g0;
  g0[0] = 1u;                                               // count=1, is_restore=0
  g0[1] = lds_addr;                                         // lds_addr (bytes)
  g0[2] = (unsigned)(gaddr & 0xFFFFFFFFull);                // global_addr[31:0]
  g0[3] = (unsigned)((gaddr >> 32) & 0x1FFFFFFull) | (2u << 30); // global_addr[56:32] | type=2
  v8i g1;
  g1[0] = 0x00030000;                                       // wg_mask=0, data_size=3 (8B)
  g1[1] = (int)((cols_q & 0xFFFFu) << 16);                  // tensor_dim0[15:0]
  g1[2] = (int)((cols_q >> 16) | ((rows & 0xFFFFu) << 16)); // tensor_dim0[31:16] | tensor_dim1[15:0]
  g1[3] = (int)((rows >> 16) | ((cols_q & 0xFFFFu) << 16)); // tensor_dim1[31:16] | tile_dim0
  g1[4] = (int)(rows & 0xFFFFu);                            // tile_dim1 | tile_dim2=0
  g1[5] = (int)cols_q;                                      // tensor_dim0_stride[31:0]
  g1[6] = 0;
  g1[7] = 0;
  v4i z4; z4[0] = 0; z4[1] = 0; z4[2] = 0; z4[3] = 0;
  v8i z8;
#pragma unroll
  for (int i = 0; i < 8; i++) z8[i] = 0;
  __builtin_amdgcn_tensor_load_to_lds(g0, g1, z4, z4, z8, 0);
}

// ---------------- GRU layer (per-wave 16 rows) ----------------
__device__ __forceinline__ void gru_layer(SMem& sm, const _Float16* xin, _Float16* hself,
                                          const float* bih, const float* bhh,
                                          int rowbase, int lm, int hi) {
  const _Float16* wih = sm.wbuf;
  const _Float16* whh = sm.wbuf + 3 * HP * KP;
  v16h ax[4], ah[4];
#pragma unroll
  for (int s = 0; s < 4; s++) {
    ax[s] = loadA(xin,   rowbase, s, lm, hi);
    ah[s] = loadA(hself, rowbase, s, lm, hi);
  }
  float hnew[7][8];
#pragma unroll
  for (int j = 0; j < 7; j++) {
    v8f aR = vzero8(), aZ = vzero8(), aXn = vzero8(), aHn = vzero8();
#pragma unroll
    for (int s = 0; s < 4; s++) aR  = wmma16(ax[s], loadB(wih, 0 * HP + 16 * j, s, lm, hi), aR);
#pragma unroll
    for (int s = 0; s < 4; s++) aR  = wmma16(ah[s], loadB(whh, 0 * HP + 16 * j, s, lm, hi), aR);
#pragma unroll
    for (int s = 0; s < 4; s++) aZ  = wmma16(ax[s], loadB(wih, 1 * HP + 16 * j, s, lm, hi), aZ);
#pragma unroll
    for (int s = 0; s < 4; s++) aZ  = wmma16(ah[s], loadB(whh, 1 * HP + 16 * j, s, lm, hi), aZ);
#pragma unroll
    for (int s = 0; s < 4; s++) aXn = wmma16(ax[s], loadB(wih, 2 * HP + 16 * j, s, lm, hi), aXn);
#pragma unroll
    for (int s = 0; s < 4; s++) aHn = wmma16(ah[s], loadB(whh, 2 * HP + 16 * j, s, lm, hi), aHn);
    int col = 16 * j + lm;                       // column within gate (0..111)
    float br  = bih[col]        + bhh[col];
    float bz  = bih[HP + col]   + bhh[HP + col];
    float bxn = bih[2*HP + col];
    float bhn = bhh[2*HP + col];
#pragma unroll
    for (int i = 0; i < 8; i++) {
      float r = sigm(aR[i] + br);
      float z = sigm(aZ[i] + bz);
      float n = tanh_f(aXn[i] + bxn + r * (aHn[i] + bhn));
      float hold = (float)hself[(rowbase + i + hi * 8) * KP + col];
      hnew[j][i] = (1.f - z) * n + z * hold;
    }
  }
#pragma unroll
  for (int j = 0; j < 7; j++) {
    int col = 16 * j + lm;
    if (col < H) {
#pragma unroll
      for (int i = 0; i < 8; i++)
        hself[(rowbase + i + hi * 8) * KP + col] = (_Float16)hnew[j][i];
    }
  }
}

// ---------------- Main persistent decoder kernel ----------------
__global__ __launch_bounds__(256) void decoder_kernel(
    const float* __restrict__ enc, const float* __restrict__ bproj,
    const float* __restrict__ bih0, const float* __restrict__ bhh0,
    const float* __restrict__ bih1, const float* __restrict__ bhh1,
    const float* __restrict__ bih2, const float* __restrict__ bhh2,
    const float* __restrict__ bout,
    const _Float16* __restrict__ ws,
    float* __restrict__ out) {
  __shared__ SMem sm;
  const int tid  = threadIdx.x;
  const int wave = tid >> 5;
  const int lane = tid & 31;
  const int lm   = lane & 15;
  const int hi   = lane >> 4;
  const int rowbase = wave * 16;
  const size_t gRow0 = (size_t)blockIdx.x * ROWS;

  // ---- prologue: stage biases, wout, x0 zero, enc->h0 (f16) ----
  for (int i = tid; i < ROWS * KP; i += 256) sm.x0[i] = (_Float16)0.f;
  for (int i = tid; i < 2160; i += 256) {
    float v = 0.f;
    if (i < HP) { if (i < H) v = bproj[i]; }
    else if (i < 2128) {
      int e = i - HP;
      int l = e / 672, r = e % 672;
      int which = r / 336, k = r % 336;
      int gate = k / HP, n = k % HP;
      const float* b = which ? (l == 0 ? bhh0 : l == 1 ? bhh1 : bhh2)
                             : (l == 0 ? bih0 : l == 1 ? bih1 : bih2);
      if (n < H) v = b[gate * H + n];
    } else {
      int j = i - 2128;
      if (j < VOCAB) v = bout[j];
    }
    sm.bias[i] = v;
  }
  {
    const _Float16* wsrc = ws + WS_WOUT_OFF;
    for (int i = tid; i < 32 * KP; i += 256) sm.wout[i] = wsrc[i];
  }
  for (int i = tid; i < ROWS * KP; i += 256) {
    int r = i >> 7, c = i & 127;
    sm.h[0][i] = (_Float16)enc[(gRow0 + (size_t)r) * LATENT + c];
  }
  // stage w_proj via TDM: 28672 B = 3584 qwords = 512 x 7
  if (wave == 0) {
    unsigned ldsa = (unsigned)(uintptr_t)&sm.wbuf[0];
    tdm_issue(ldsa, (unsigned long long)(uintptr_t)(ws + WS_WPROJ_OFF), 512u, 7u);
    __builtin_amdgcn_s_wait_tensorcnt(0);
  }
  __syncthreads();

  // ---- x0 = enc @ wproj.T + bproj (f16 into sm.x0) ----
  {
    v16h ae[4];
#pragma unroll
    for (int s = 0; s < 4; s++) ae[s] = loadA(sm.h[0], rowbase, s, lm, hi);
#pragma unroll
    for (int j = 0; j < 7; j++) {
      v8f acc = vzero8();
#pragma unroll
      for (int s = 0; s < 4; s++) acc = wmma16(ae[s], loadB(sm.wbuf, 16 * j, s, lm, hi), acc);
      int col = 16 * j + lm;
      if (col < H) {
        float b = sm.bias[col];
#pragma unroll
        for (int i = 0; i < 8; i++)
          sm.x0[(rowbase + i + hi * 8) * KP + col] = (_Float16)(acc[i] + b);
      }
    }
  }
  __syncthreads();
  // zero all hidden states (also restores K padding of h0)
  {
    _Float16* hall = &sm.h[0][0];
    for (int i = tid; i < 3 * ROWS * KP; i += 256) hall[i] = (_Float16)0.f;
  }

  // ---- time loop ----
#pragma unroll 1
  for (int t = 0; t < TSTEPS; t++) {
#pragma unroll 1
    for (int l = 0; l < 3; l++) {
      __syncthreads();                 // all waves done with previous wbuf contents
      if (wave == 0) {
        unsigned ldsa = (unsigned)(uintptr_t)&sm.wbuf[0];
        unsigned long long ga =
            (unsigned long long)(uintptr_t)(ws + WS_BIG_OFF + (size_t)l * 2 * WS_BIG_ONE);
        // 172032 B = 21504 qwords = 672 x 32
        tdm_issue(ldsa, ga, 672u, 32u);
        __builtin_amdgcn_s_wait_tensorcnt(0);
      }
      __syncthreads();                 // wbuf now holds (wih_l | whh_l)
      const _Float16* xin = (l == 0) ? sm.x0 : sm.h[l - 1];
      const float* bih = &sm.bias[HP + l * 672];
      gru_layer(sm, xin, sm.h[l], bih, bih + 336, rowbase, lm, hi);
    }
    // ---- logits = h2 @ wout.T + bout ----
    v16h a2[4];
#pragma unroll
    for (int s = 0; s < 4; s++) a2[s] = loadA(sm.h[2], rowbase, s, lm, hi);
#pragma unroll
    for (int jo = 0; jo < 2; jo++) {
      v8f acc = vzero8();
#pragma unroll
      for (int s = 0; s < 4; s++) acc = wmma16(a2[s], loadB(sm.wout, 16 * jo, s, lm, hi), acc);
      int v = 16 * jo + lm;
      if (v < VOCAB) {
        float bo = sm.bias[2128 + v];
#pragma unroll
        for (int i = 0; i < 8; i++) {
          size_t row = gRow0 + (size_t)(rowbase + i + hi * 8);
          out[(row * TSTEPS + (size_t)t) * VOCAB + (size_t)v] = acc[i] + bo;
        }
      }
    }
  }
}

// ---------------- Host launcher ----------------
extern "C" void kernel_launch(void* const* d_in, const int* in_sizes, int n_in,
                              void* d_out, int out_size, void* d_ws, size_t ws_size,
                              hipStream_t stream) {
  (void)in_sizes; (void)n_in; (void)out_size; (void)ws_size;
  const float* enc   = (const float*)d_in[0];
  const float* wproj = (const float*)d_in[1];
  const float* bproj = (const float*)d_in[2];
  const float* wih0  = (const float*)d_in[3];
  const float* whh0  = (const float*)d_in[4];
  const float* bih0  = (const float*)d_in[5];
  const float* bhh0  = (const float*)d_in[6];
  const float* wih1  = (const float*)d_in[7];
  const float* whh1  = (const float*)d_in[8];
  const float* bih1  = (const float*)d_in[9];
  const float* bhh1  = (const float*)d_in[10];
  const float* wih2  = (const float*)d_in[11];
  const float* whh2  = (const float*)d_in[12];
  const float* bih2  = (const float*)d_in[13];
  const float* bhh2  = (const float*)d_in[14];
  const float* wout  = (const float*)d_in[15];
  const float* bout  = (const float*)d_in[16];
  _Float16* ws = (_Float16*)d_ws;
  float* out = (float*)d_out;

  const int totalW = HP * KP + 6 * 3 * HP * KP + 32 * KP;
  prep_weights<<<(totalW + 255) / 256, 256, 0, stream>>>(
      wproj, wih0, whh0, wih1, whh1, wih2, whh2, wout, ws);
  decoder_kernel<<<B_TOT / ROWS, 256, 0, stream>>>(
      enc, bproj, bih0, bhh0, bih1, bhh1, bih2, bhh2, bout, ws, out);
}
